// GnBlock_46067819217346
// MI455X (gfx1250) — matrix operand order repaired and checked by reading, not compiled, not run on hardware
//
#include <hip/hip_runtime.h>

#define DD 128

typedef __attribute__((ext_vector_type(16))) __bf16 v16bf;
typedef __attribute__((ext_vector_type(8)))  float  v8f;

__device__ __forceinline__ float silu_f(float v) { return v / (1.f + __expf(-v)); }

// Per-wave LDS ordering fence: LDS ops from one wave complete in order; this
// just stops the compiler from reordering and drains DScnt.
__device__ __forceinline__ void wave_lds_fence() {
  __builtin_amdgcn_wave_barrier();
  asm volatile("s_wait_dscnt 0" ::: "memory");
  __builtin_amdgcn_wave_barrier();
}

// Fill 8 consecutive A-fragment elements with LN-affine-transformed bf16 values.
#define LN8(A, OFF, V0, V1, GA, GB, BA, BB, MU, RS) do {              \
    (A)[(OFF)+0] = (__bf16)((((V0).x-(MU))*(RS))*(GA).x + (BA).x);    \
    (A)[(OFF)+1] = (__bf16)((((V0).y-(MU))*(RS))*(GA).y + (BA).y);    \
    (A)[(OFF)+2] = (__bf16)((((V0).z-(MU))*(RS))*(GA).z + (BA).z);    \
    (A)[(OFF)+3] = (__bf16)((((V0).w-(MU))*(RS))*(GA).w + (BA).w);    \
    (A)[(OFF)+4] = (__bf16)((((V1).x-(MU))*(RS))*(GB).x + (BB).x);    \
    (A)[(OFF)+5] = (__bf16)((((V1).y-(MU))*(RS))*(GB).y + (BB).y);    \
    (A)[(OFF)+6] = (__bf16)((((V1).z-(MU))*(RS))*(GB).z + (BB).z);    \
    (A)[(OFF)+7] = (__bf16)((((V1).w-(MU))*(RS))*(GB).w + (BB).w);    \
  } while (0)

// Hidden 128->128 layer: LN(sh row) @ Wt (bias handled by caller).
// sh: per-wave [16][DD+4] f32 activations (row-major). Wt: bf16 [128][128] (N-major).
// Produces acc[8] 16x16 f32 C-tiles covering all 128 output columns.
__device__ __forceinline__ void dense128_layer(
    const float (*sh)[DD + 4], int l15, int half,
    const float* __restrict__ g, const float* __restrict__ bln,
    const __bf16* __restrict__ Wt, v8f acc[8])
{
  // LN stats: lanes L and L+16 cooperate on row l15.
  float s = 0.f, q = 0.f;
  #pragma unroll
  for (int k = 0; k < 64; k += 4) {
    const float4 v = *(const float4*)&sh[l15][half * 64 + k];
    s += v.x + v.y + v.z + v.w;
    q += v.x*v.x + v.y*v.y + v.z*v.z + v.w*v.w;
  }
  s += __shfl_xor(s, 16);
  q += __shfl_xor(q, 16);
  const float mu = s * (1.f / 128.f);
  const float rs = rsqrtf(q * (1.f / 128.f) - mu * mu + 1e-5f);

  v8f zero = {};
  #pragma unroll
  for (int t = 0; t < 8; ++t) acc[t] = zero;

  for (int k0 = 0; k0 < DD; k0 += 32) {
    v16bf afrag;
    #pragma unroll
    for (int gi = 0; gi < 2; ++gi) {
      const int kb = k0 + gi * 16 + half * 8;
      const float4 v0 = *(const float4*)&sh[l15][kb];
      const float4 v1 = *(const float4*)&sh[l15][kb + 4];
      const float4 ga = *(const float4*)(g + kb);
      const float4 gb = *(const float4*)(g + kb + 4);
      const float4 ba = *(const float4*)(bln + kb);
      const float4 bb = *(const float4*)(bln + kb + 4);
      LN8(afrag, gi * 8, v0, v1, ga, gb, ba, bb, mu, rs);
    }
    const __bf16* wb = Wt + (size_t)l15 * DD + k0 + half * 8;
    #pragma unroll
    for (int t = 0; t < 8; ++t) {
      typedef __attribute__((ext_vector_type(8))) __bf16 v8bf_t;
      const v8bf_t blo = *(const v8bf_t*)(wb + (size_t)(t * 16) * DD);
      const v8bf_t bhi = *(const v8bf_t*)(wb + (size_t)(t * 16) * DD + 16);
      const v16bf bfrag = __builtin_shufflevector(blo, bhi, 0,1,2,3,4,5,6,7,8,9,10,11,12,13,14,15);
      acc[t] = __builtin_amdgcn_wmma_f32_16x16x32_bf16(
          false, afrag, false, bfrag, (short)0, acc[t], false, false);
    }
  }
}

// 3-way segment select for the concatenated edge input [x[src] | x[dst] | edge_attr].
__device__ __forceinline__ const float* seg3(const float* s0, const float* s1,
                                             const float* s2, int kk) {
  return kk < DD ? (s0 + kk) : (kk < 2 * DD ? (s1 + (kk - DD)) : (s2 + (kk - 2 * DD)));
}

__global__ void __launch_bounds__(128, 4) gn_edge_kernel(
    const float* __restrict__ x, const float* __restrict__ ea,
    const int* __restrict__ src, const int* __restrict__ dst,
    const float* __restrict__ g0, const float* __restrict__ b0,
    const __bf16* __restrict__ W1t, const float* __restrict__ b1,
    const float* __restrict__ g1, const float* __restrict__ b1n,
    const __bf16* __restrict__ W2t, const float* __restrict__ b2,
    const float* __restrict__ g2, const float* __restrict__ b2n,
    const __bf16* __restrict__ W3t, const float* __restrict__ b3,
    float* agg, float* edge_new, int E)
{
  __shared__ float S[4][16][DD + 4];
  const int wave = threadIdx.x >> 5;
  const int lane = threadIdx.x & 31;
  const int half = lane >> 4;
  const int l15  = lane & 15;
  float (*sh)[DD + 4] = S[wave];

  // One 16-edge tile per wave (no loop: keeps weight fragments from being
  // hoisted/spilled across an outer loop).
  const int tile = blockIdx.x * 4 + wave;
  const int ntiles = (E + 15) >> 4;
  if (tile >= ntiles) return;   // wave-uniform

  const int e  = tile * 16 + l15;
  const int ec = e < E ? e : (E - 1);
  const float* seg0 = x  + (size_t)src[ec] * DD;
  const float* seg1 = x  + (size_t)dst[ec] * DD;
  const float* seg2 = ea + (size_t)ec * DD;

  // ---- LN stats over 384-dim input (2 lanes per row) ----
  float sum = 0.f, sq = 0.f;
  {
    const int base = half * 192;
    #pragma unroll
    for (int k = 0; k < 192; k += 4) {
      const float4 v = *(const float4*)seg3(seg0, seg1, seg2, base + k);
      sum += v.x + v.y + v.z + v.w;
      sq  += v.x*v.x + v.y*v.y + v.z*v.z + v.w*v.w;
    }
  }
  sum += __shfl_xor(sum, 16);
  sq  += __shfl_xor(sq, 16);
  const float mu = sum * (1.f / 384.f);
  const float rs = rsqrtf(sq * (1.f / 384.f) - mu * mu + 1e-5f);

  // ---- Layer 1: [16x384] @ [384x128] ----
  v8f acc[8];
  {
    v8f zero = {};
    #pragma unroll
    for (int t = 0; t < 8; ++t) acc[t] = zero;
  }
  for (int k0 = 0; k0 < 3 * DD; k0 += 32) {
    v16bf afrag;
    #pragma unroll
    for (int gi = 0; gi < 2; ++gi) {
      const int kb = k0 + gi * 16 + half * 8;
      const float* p = seg3(seg0, seg1, seg2, kb);
      const float4 v0 = *(const float4*)(p);
      const float4 v1 = *(const float4*)(p + 4);
      const float4 ga = *(const float4*)(g0 + kb);
      const float4 gb = *(const float4*)(g0 + kb + 4);
      const float4 ba = *(const float4*)(b0 + kb);
      const float4 bb = *(const float4*)(b0 + kb + 4);
      LN8(afrag, gi * 8, v0, v1, ga, gb, ba, bb, mu, rs);
    }
    const __bf16* wb = W1t + (size_t)l15 * (3 * DD) + k0 + half * 8;
    #pragma unroll
    for (int t = 0; t < 8; ++t) {
      typedef __attribute__((ext_vector_type(8))) __bf16 v8bf_t;
      const v8bf_t blo = *(const v8bf_t*)(wb + (size_t)(t * 16) * (3 * DD));
      const v8bf_t bhi = *(const v8bf_t*)(wb + (size_t)(t * 16) * (3 * DD) + 16);
      const v16bf bfrag = __builtin_shufflevector(blo, bhi, 0,1,2,3,4,5,6,7,8,9,10,11,12,13,14,15);
      acc[t] = __builtin_amdgcn_wmma_f32_16x16x32_bf16(
          false, afrag, false, bfrag, (short)0, acc[t], false, false);
    }
  }

  // bias + SiLU -> LDS
  wave_lds_fence();
  #pragma unroll
  for (int t = 0; t < 8; ++t) {
    const int col = t * 16 + l15;
    const float bv = b1[col];
    #pragma unroll
    for (int r = 0; r < 8; ++r)
      sh[r + half * 8][col] = silu_f(acc[t][r] + bv);
  }
  wave_lds_fence();

  // ---- Layer 2 ----
  dense128_layer(sh, l15, half, g1, b1n, W2t, acc);
  wave_lds_fence();
  #pragma unroll
  for (int t = 0; t < 8; ++t) {
    const int col = t * 16 + l15;
    const float bv = b2[col];
    #pragma unroll
    for (int r = 0; r < 8; ++r)
      sh[r + half * 8][col] = silu_f(acc[t][r] + bv);
  }
  wave_lds_fence();

  // ---- Layer 3 (no activation) ----
  dense128_layer(sh, l15, half, g2, b2n, W3t, acc);

  // ---- Epilogue: edge_new = edge_attr + 0.1*e_out ; atomic scatter into agg ----
  #pragma unroll
  for (int r = 0; r < 8; ++r) {
    const int er = tile * 16 + r + half * 8;
    if (er < E) {
      const int dn = dst[er];
      const float* ear = ea + (size_t)er * DD;
      float* enr = edge_new + (size_t)er * DD;
      float* agr = agg + (size_t)dn * DD;
      #pragma unroll
      for (int t = 0; t < 8; ++t) {
        const int col = t * 16 + l15;
        const float v = acc[t][r] + b3[col];
        enr[col] = ear[col] + 0.1f * v;
        atomicAdd(agr + col, v);
      }
    }
  }
}

__global__ void __launch_bounds__(128, 4) gn_node_kernel(
    const float* __restrict__ x, float* xa /* agg in, x_new out */,
    const float* __restrict__ g0, const float* __restrict__ b0,
    const __bf16* __restrict__ W1t, const float* __restrict__ b1,
    const float* __restrict__ g1, const float* __restrict__ b1n,
    const __bf16* __restrict__ W2t, const float* __restrict__ b2,
    const float* __restrict__ g2, const float* __restrict__ b2n,
    const __bf16* __restrict__ W3t, const float* __restrict__ b3,
    int N)
{
  __shared__ float S[4][16][DD + 4];
  const int wave = threadIdx.x >> 5;
  const int lane = threadIdx.x & 31;
  const int half = lane >> 4;
  const int l15  = lane & 15;
  float (*sh)[DD + 4] = S[wave];

  const int tile = blockIdx.x * 4 + wave;
  const int ntiles = (N + 15) >> 4;
  if (tile >= ntiles) return;   // wave-uniform

  const int i  = tile * 16 + l15;
  const int ic = i < N ? i : (N - 1);
  const float* seg0 = x  + (size_t)ic * DD;
  const float* seg1 = xa + (size_t)ic * DD;   // agg row

  // ---- LN stats over 256-dim input ----
  float sum = 0.f, sq = 0.f;
  {
    const int base = half * 128;
    #pragma unroll
    for (int k = 0; k < 128; k += 4) {
      const int kk = base + k;
      const float* p = kk < DD ? (seg0 + kk) : (seg1 + (kk - DD));
      const float4 v = *(const float4*)p;
      sum += v.x + v.y + v.z + v.w;
      sq  += v.x*v.x + v.y*v.y + v.z*v.z + v.w*v.w;
    }
  }
  sum += __shfl_xor(sum, 16);
  sq  += __shfl_xor(sq, 16);
  const float mu = sum * (1.f / 256.f);
  const float rs = rsqrtf(sq * (1.f / 256.f) - mu * mu + 1e-5f);

  // ---- Layer 1: [16x256] @ [256x128] ----
  v8f acc[8];
  {
    v8f zero = {};
    #pragma unroll
    for (int t = 0; t < 8; ++t) acc[t] = zero;
  }
  for (int k0 = 0; k0 < 2 * DD; k0 += 32) {
    v16bf afrag;
    #pragma unroll
    for (int gi = 0; gi < 2; ++gi) {
      const int kb = k0 + gi * 16 + half * 8;
      const float* p = kb < DD ? (seg0 + kb) : (seg1 + (kb - DD));
      const float4 v0 = *(const float4*)(p);
      const float4 v1 = *(const float4*)(p + 4);
      const float4 ga = *(const float4*)(g0 + kb);
      const float4 gb = *(const float4*)(g0 + kb + 4);
      const float4 ba = *(const float4*)(b0 + kb);
      const float4 bb = *(const float4*)(b0 + kb + 4);
      LN8(afrag, gi * 8, v0, v1, ga, gb, ba, bb, mu, rs);
    }
    const __bf16* wb = W1t + (size_t)l15 * (2 * DD) + k0 + half * 8;
    #pragma unroll
    for (int t = 0; t < 8; ++t) {
      typedef __attribute__((ext_vector_type(8))) __bf16 v8bf_t;
      const v8bf_t blo = *(const v8bf_t*)(wb + (size_t)(t * 16) * (2 * DD));
      const v8bf_t bhi = *(const v8bf_t*)(wb + (size_t)(t * 16) * (2 * DD) + 16);
      const v16bf bfrag = __builtin_shufflevector(blo, bhi, 0,1,2,3,4,5,6,7,8,9,10,11,12,13,14,15);
      acc[t] = __builtin_amdgcn_wmma_f32_16x16x32_bf16(
          false, afrag, false, bfrag, (short)0, acc[t], false, false);
    }
  }

  wave_lds_fence();
  #pragma unroll
  for (int t = 0; t < 8; ++t) {
    const int col = t * 16 + l15;
    const float bv = b1[col];
    #pragma unroll
    for (int r = 0; r < 8; ++r)
      sh[r + half * 8][col] = silu_f(acc[t][r] + bv);
  }
  wave_lds_fence();

  dense128_layer(sh, l15, half, g1, b1n, W2t, acc);
  wave_lds_fence();
  #pragma unroll
  for (int t = 0; t < 8; ++t) {
    const int col = t * 16 + l15;
    const float bv = b2[col];
    #pragma unroll
    for (int r = 0; r < 8; ++r)
      sh[r + half * 8][col] = silu_f(acc[t][r] + bv);
  }
  wave_lds_fence();

  dense128_layer(sh, l15, half, g2, b2n, W3t, acc);

  // ---- Epilogue: x_new = x + 0.1*n_out (overwrites this tile's agg rows) ----
  #pragma unroll
  for (int r = 0; r < 8; ++r) {
    const int ir = tile * 16 + r + half * 8;
    if (ir < N) {
      const float* xr = x + (size_t)ir * DD;
      float* outr = xa + (size_t)ir * DD;
      #pragma unroll
      for (int t = 0; t < 8; ++t) {
        const int col = t * 16 + l15;
        const float v = acc[t][r] + b3[col];
        outr[col] = xr[col] + 0.1f * v;
      }
    }
  }
}

// Transpose + convert f32 W[K][N] -> bf16 Wt[N][K]
__global__ void wt_convert_kernel(const float* __restrict__ W, __bf16* __restrict__ Wt,
                                  int K, int Nn) {
  const int idx = blockIdx.x * blockDim.x + threadIdx.x;
  if (idx >= K * Nn) return;
  const int k = idx / Nn;
  const int n = idx % Nn;
  Wt[(size_t)n * K + k] = (__bf16)W[idx];
}

__global__ void zero_f32_kernel(float* __restrict__ p, int n) {
  const int idx = blockIdx.x * blockDim.x + threadIdx.x;
  if (idx < n) p[idx] = 0.f;
}

extern "C" void kernel_launch(void* const* d_in, const int* in_sizes, int n_in,
                              void* d_out, int out_size, void* d_ws, size_t ws_size,
                              hipStream_t stream) {
  (void)n_in; (void)out_size; (void)ws_size;

  const float* x  = (const float*)d_in[0];
  const float* ea = (const float*)d_in[1];
  const int*   ei = (const int*)d_in[2];
  const int N = in_sizes[0] / DD;
  const int E = in_sizes[1] / DD;

  // Param dicts flattened in insertion order: g0,b0,W1,b1,g1,b1n,W2,b2,g2,b2n,W3,b3
  const float* ep[12];
  const float* np[12];
  for (int i = 0; i < 12; ++i) ep[i] = (const float*)d_in[3 + i];
  for (int i = 0; i < 12; ++i) np[i] = (const float*)d_in[15 + i];

  // Workspace: transposed bf16 weights (~295 KB total)
  __bf16* ws = (__bf16*)d_ws;
  __bf16* eW1t = ws;                 // 128 x 384
  __bf16* eW2t = ws + 49152;         // 128 x 128
  __bf16* eW3t = ws + 65536;         // 128 x 128
  __bf16* nW1t = ws + 81920;         // 128 x 256
  __bf16* nW2t = ws + 114688;        // 128 x 128
  __bf16* nW3t = ws + 131072;        // 128 x 128

  wt_convert_kernel<<<(384 * 128 + 255) / 256, 256, 0, stream>>>(ep[2],  eW1t, 384, 128);
  wt_convert_kernel<<<(128 * 128 + 255) / 256, 256, 0, stream>>>(ep[6],  eW2t, 128, 128);
  wt_convert_kernel<<<(128 * 128 + 255) / 256, 256, 0, stream>>>(ep[10], eW3t, 128, 128);
  wt_convert_kernel<<<(256 * 128 + 255) / 256, 256, 0, stream>>>(np[2],  nW1t, 256, 128);
  wt_convert_kernel<<<(128 * 128 + 255) / 256, 256, 0, stream>>>(np[6],  nW2t, 128, 128);
  wt_convert_kernel<<<(128 * 128 + 255) / 256, 256, 0, stream>>>(np[10], nW3t, 128, 128);

  float* xnew     = (float*)d_out;              // doubles as agg accumulator
  float* edge_new = xnew + (size_t)N * DD;

  zero_f32_kernel<<<(N * DD + 255) / 256, 256, 0, stream>>>(xnew, N * DD);

  const int* srcI = ei;
  const int* dstI = ei + E;
  const int etiles = (E + 15) / 16;
  gn_edge_kernel<<<(etiles + 3) / 4, 128, 0, stream>>>(
      x, ea, srcI, dstI,
      ep[0], ep[1], eW1t, ep[3], ep[4], ep[5], eW2t, ep[7], ep[8], ep[9], eW3t, ep[11],
      xnew, edge_new, E);

  const int ntiles = (N + 15) / 16;
  gn_node_kernel<<<(ntiles + 3) / 4, 128, 0, stream>>>(
      x, xnew,
      np[0], np[1], nW1t, np[3], np[4], np[5], nW2t, np[7], np[8], np[9], nW3t, np[11],
      N);
}
